// Multi_Omics_Cooperative_Transformer_42966852829495
// MI455X (gfx1250) — matrix operand order repaired and checked by reading, not compiled, not run on hardware
//
#include <hip/hip_runtime.h>
#include <hip/hip_bf16.h>
#include <math.h>

typedef __attribute__((ext_vector_type(16))) __bf16 v16bf;
typedef __attribute__((ext_vector_type(8)))  float  v8f;
typedef int v4i_ __attribute__((vector_size(16)));

#define LDPAD 112   // 100 cols padded to 7 WMMA tiles of 16
#define NROWS 6144

// ---------------- async global->LDS copy (CDNA5), with sync fallback --------
#if defined(__AMDGCN__) && __has_builtin(__builtin_amdgcn_global_load_async_to_lds_b128)
#define ASYNC_CP 1
#else
#define ASYNC_CP 0
#endif

#if ASYNC_CP
# if __has_builtin(__builtin_amdgcn_s_wait_asynccnt)
#  define ASYNC_WAIT(n) __builtin_amdgcn_s_wait_asynccnt(n)
# else
#  define ASYNC_WAIT(n) asm volatile("s_wait_asynccnt %0" :: "i"(n) : "memory")
# endif
#else
# define ASYNC_WAIT(n) ((void)0)
#endif

__device__ __forceinline__ void async_cp16(const float* g, float* l) {
#if ASYNC_CP
    typedef __attribute__((address_space(1))) v4i_ GV;   // global 128-bit payload
    typedef __attribute__((address_space(3))) v4i_ LV;   // LDS 128-bit payload
    __builtin_amdgcn_global_load_async_to_lds_b128((GV*)g, (LV*)l, 0, 0);
#else
    *(uint4*)l = *(const uint4*)g;
#endif
}

__device__ __forceinline__ float wave_sum32(float v) {
    for (int o = 16; o; o >>= 1) v += __shfl_xor(v, o, 32);
    return v;
}

// ---------------------------------------------------------------------------
// C[N,100(pad112)] = A[N,K] @ W[K,100] + bias, bf16 WMMA, f32 accum.
// Block = 128 threads (4 waves); 64x16 C tile. A tiles are double-buffered
// f32 in LDS, filled with async global->LDS b128 copies one K-chunk ahead
// (each wave stages exactly the 16 rows it consumes); bf16 conversion happens
// at fragment build. K tail uses a guarded synchronous path.
// ---------------------------------------------------------------------------
__global__ __launch_bounds__(128) void gemm_bias_kernel(
    const float* __restrict__ A, int lda,
    const float* __restrict__ W, int ldw,
    const float* __restrict__ bias,
    float* __restrict__ C, int ldc,
    int K, int M, __bf16* __restrict__ Cbf)
{
    __shared__ __align__(16) float  Af[2][64][32];      // 16 KB, double buffered
    __shared__ __align__(16) __bf16 BsubT[2][16][32];   // BsubT[b][n][k] = W[k][c0+n]
    const int tid = threadIdx.x, wv = tid >> 5, l = tid & 31;
    const int c0 = blockIdx.x * 16, r0 = blockIdx.y * 64;
    const int ml = l & 15;
    const int hk = (l & 16) ? 8 : 0;       // unused; A-frag K base is bk too
    const int bk = (l >> 4) << 4;          // B-frag K base within chunk
    const bool col_edge = (c0 + 16 > M);

    // per-lane async mapping: wave wv stages rows [wv*16, wv*16+16)
    const int arow = wv * 16 + (l >> 1);   // row within 64-row tile
    const int akb  = (l & 1) << 4;         // k base: 0 or 16 (4 float4 each)

    auto issueA = [&](int kk, int buf) {
        const float* src = A + (size_t)(r0 + arow) * lda + kk + akb;
        float* dst = &Af[buf][arow][akb];
        #pragma unroll
        for (int p = 0; p < 4; ++p) async_cp16(src + 4 * p, dst + 4 * p);
    };
    auto stageA_tail = [&](int kk, int buf) {   // guarded sync path (K tail)
        for (int e = tid; e < 2048; e += 128) {
            int rr = e >> 5, k = e & 31;
            float v = (kk + k < K) ? A[(size_t)(r0 + rr) * lda + kk + k] : 0.0f;
            Af[buf][rr][k] = v;
        }
    };

    const int nchunks = (K + 31) >> 5;
    if (K >= 32) issueA(0, 0); else stageA_tail(0, 0);

    v8f acc = {};
    for (int c = 0; c < nchunks; ++c) {
        const int kk = c << 5, buf = c & 1;
        // ---- stage W tile for this chunk (sync, tiny / L2 resident) ----
        if (kk + 32 <= K && !col_edge) {
            const int k = tid >> 2, q4 = (tid & 3) << 2;
            float4 f = *(const float4*)(W + (size_t)(kk + k) * ldw + c0 + q4);
            BsubT[buf][q4 + 0][k] = (__bf16)f.x;
            BsubT[buf][q4 + 1][k] = (__bf16)f.y;
            BsubT[buf][q4 + 2][k] = (__bf16)f.z;
            BsubT[buf][q4 + 3][k] = (__bf16)f.w;
        } else {
            for (int e = tid; e < 512; e += 128) {
                int nn = e >> 5, k = e & 31;
                int col = c0 + nn;
                float v = (kk + k < K && col < M) ? W[(size_t)(kk + k) * ldw + col] : 0.0f;
                BsubT[buf][nn][k] = (__bf16)v;
            }
        }
        // ---- stream next A chunk one iteration ahead ----
        const int kn = kk + 32;
        if (kn < K) {
            if (kn + 32 <= K) { issueA(kn, buf ^ 1); ASYNC_WAIT(4); }
            else              { stageA_tail(kn, buf ^ 1); ASYNC_WAIT(0); }
        } else {
            ASYNC_WAIT(0);
        }
        __syncthreads();
        // ---- fragments (f32 LDS -> packed bf16 cvt) + WMMA ----
        v16bf a, b;
        const float* ar = &Af[buf][(wv << 4) + ml][0];
        #pragma unroll
        for (int i = 0; i < 16; ++i) a[i] = (__bf16)ar[bk + i];
        #pragma unroll
        for (int i = 0; i < 16; ++i) b[i] = BsubT[buf][ml][bk + i];
        acc = __builtin_amdgcn_wmma_f32_16x16x32_bf16(false, a, false, b, (short)0, acc,
                                                      false, false);
        __syncthreads();
    }
    const int col = c0 + ml;
    const int hr = (l >> 4) << 3;
    #pragma unroll
    for (int r = 0; r < 8; ++r) {
        int row = r0 + wv * 16 + r + hr;
        float v = (col < M) ? (acc[r] + bias[col]) : 0.0f;
        C[(size_t)row * ldc + col] = v;
        if (Cbf) Cbf[(size_t)row * ldc + col] = (__bf16)v;
    }
}

// ---------------------------------------------------------------------------
// BatchNorm training-mode batch statistics over 100 columns (ld = 112).
// ---------------------------------------------------------------------------
__global__ __launch_bounds__(256) void bn_stats_kernel(
    const float* __restrict__ x, int nrows,
    float* __restrict__ mean, float* __restrict__ inv)
{
    __shared__ float ls[256], lq[256];
    const int c = blockIdx.x, t = threadIdx.x;
    float s = 0.f, q = 0.f;
    for (int r = t; r < nrows; r += 256) {
        float v = x[(size_t)r * LDPAD + c];
        s += v; q += v * v;
    }
    ls[t] = s; lq[t] = q;
    __syncthreads();
    for (int o = 128; o; o >>= 1) {
        if (t < o) { ls[t] += ls[t + o]; lq[t] += lq[t + o]; }
        __syncthreads();
    }
    if (t == 0) {
        float m = ls[0] / (float)nrows;
        float var = lq[0] / (float)nrows - m * m;
        mean[c] = m;
        inv[c] = rsqrtf(var + 1e-5f);
    }
}

__global__ __launch_bounds__(256) void bn_apply_kernel(
    float* __restrict__ x, const float* __restrict__ mean, const float* __restrict__ inv,
    const float* __restrict__ g, const float* __restrict__ be, int nrows, int do_tanh)
{
    int idx = blockIdx.x * 256 + threadIdx.x;
    if (idx >= nrows * 100) return;
    int r = idx / 100, c = idx - r * 100;
    float v = g[c] * (x[(size_t)r * LDPAD + c] - mean[c]) * inv[c] + be[c];
    x[(size_t)r * LDPAD + c] = do_tanh ? tanhf(v) : v;
}

// ---------------------------------------------------------------------------
// Two-head 2-token attention + fuse + sign-sqrt + l2norm. One wave per sample.
// ---------------------------------------------------------------------------
__global__ __launch_bounds__(128) void token_attn_fuse_kernel(
    const float* __restrict__ gf, const float* __restrict__ mf,
    const float* __restrict__ wq1, const float* __restrict__ bq1,
    const float* __restrict__ wv1, const float* __restrict__ bv1,
    const float* __restrict__ wq2, const float* __restrict__ bq2,
    const float* __restrict__ wv2, const float* __restrict__ bv2,
    const float* __restrict__ wfuse, const float* __restrict__ bfuse,
    float* __restrict__ fused_pre)
{
    __shared__ float x0s[4][100], x1s[4][100], cats[4][100];
    const int wv = threadIdx.x >> 5, l = threadIdx.x & 31;
    const int n = blockIdx.x * 4 + wv;

    for (int c = l; c < 100; c += 32) {
        x0s[wv][c] = gf[(size_t)n * LDPAD + c];
        x1s[wv][c] = mf[(size_t)n * LDPAD + c];
    }
    __syncthreads();

    for (int h = 0; h < 2; ++h) {
        const float* wq = h ? wq2 : wq1;  const float* bq = h ? bq2 : bq1;
        const float* wvp = h ? wv2 : wv1; const float* bv = h ? bv2 : bv1;
        float V0[2], V1[2];
        float p00 = 0.f, p01 = 0.f, p11 = 0.f;
        for (int rep = 0; rep < 2; ++rep) {
            int d = l + 32 * rep;
            float q0 = 0.f, q1 = 0.f, v0 = 0.f, v1 = 0.f;
            if (d < 50) {
                q0 = bq[d]; q1 = bq[d]; v0 = bv[d]; v1 = bv[d];
                for (int k = 0; k < 100; ++k) {
                    float wk = wq[k * 50 + d], wk2 = wvp[k * 50 + d];
                    q0 += x0s[wv][k] * wk;  q1 += x1s[wv][k] * wk;
                    v0 += x0s[wv][k] * wk2; v1 += x1s[wv][k] * wk2;
                }
            }
            V0[rep] = v0; V1[rep] = v1;
            p00 += q0 * q0; p01 += q0 * q1; p11 += q1 * q1;
        }
        p00 = wave_sum32(p00); p01 = wave_sum32(p01); p11 = wave_sum32(p11);
        const float scl = 0.14142135623730951f;  // 1/sqrt(50)
        float s00 = p00 * scl, s01 = p01 * scl, s11 = p11 * scl;
        float m0 = fmaxf(s00, s01);
        float e00 = expf(s00 - m0), e01 = expf(s01 - m0);
        float a00 = e00 / (e00 + e01), a01 = e01 / (e00 + e01);
        float m1 = fmaxf(s01, s11);
        float e10 = expf(s01 - m1), e11 = expf(s11 - m1);
        float a10 = e10 / (e10 + e11), a11 = e11 / (e10 + e11);
        for (int rep = 0; rep < 2; ++rep) {
            int d = l + 32 * rep;
            if (d < 50) {
                cats[wv][d]      = a00 * V0[rep] + a01 * V1[rep];
                cats[wv][50 + d] = a10 * V0[rep] + a11 * V1[rep];
            }
        }
        __syncthreads();
        float t[2]; float ss = 0.f;
        for (int rep = 0; rep < 2; ++rep) {
            int d = l + 32 * rep;
            float tv = 0.f;
            if (d < 50) {
                tv = bfuse[d];
                for (int k = 0; k < 100; ++k) tv += cats[wv][k] * wfuse[k * 50 + d];
                tv = copysignf(sqrtf(fabsf(tv)), tv);   // sign(x)*sqrt(|x|)
            }
            t[rep] = tv; ss += tv * tv;
        }
        ss = wave_sum32(ss);
        float invn = 1.0f / fmaxf(sqrtf(ss), 1e-12f);
        for (int rep = 0; rep < 2; ++rep) {
            int d = l + 32 * rep;
            if (d < 50) fused_pre[(size_t)n * LDPAD + 50 * h + d] = t[rep] * invn;
        }
        __syncthreads();
    }
    if (l < 12) fused_pre[(size_t)n * LDPAD + 100 + l] = 0.0f;
}

// Row l2-normalize (first 100 cols, ld 112) in place. One wave per row.
__global__ __launch_bounds__(128) void l2norm_rows_kernel(float* __restrict__ x)
{
    const int wv = threadIdx.x >> 5, l = threadIdx.x & 31;
    const int row = blockIdx.x * 4 + wv;
    float vals[4]; float ss = 0.f;
    #pragma unroll
    for (int rep = 0; rep < 4; ++rep) {
        int c = l + 32 * rep;
        float v = (c < 100) ? x[(size_t)row * LDPAD + c] : 0.f;
        vals[rep] = v; ss += v * v;
    }
    ss = wave_sum32(ss);
    float inv = 1.0f / fmaxf(sqrtf(ss), 1e-12f);
    #pragma unroll
    for (int rep = 0; rep < 4; ++rep) {
        int c = l + 32 * rep;
        if (c < 100) x[(size_t)row * LDPAD + c] = vals[rep] * inv;
    }
}

// vt[c][row] = (bf16)vn[row][c]  (B-operand-transposed bf16 copy of l2n(v)).
__global__ __launch_bounds__(256) void build_vnT_kernel(
    const float* __restrict__ vn, __bf16* __restrict__ vt, int N)
{
    const int c = blockIdx.y;
    const int row = blockIdx.x * 256 + threadIdx.x;
    vt[(size_t)c * N + row] = (__bf16)vn[(size_t)row * LDPAD + c];
}

// ---------------------------------------------------------------------------
// Fused graph attention (flash style): per block 64 rows; stream j in tiles of
// 32 cols. qb ([N,112] bf16) / vt ([112,N] bf16) staged with 16B LDS copies;
// the S / S_Trans HBM stream (302 MB) is double-buffered via async
// global->LDS copies issued one j-tile ahead (each wave stages the 16 rows it
// consumes, so per-wave s_wait_asynccnt + barrier is sufficient).
// ---------------------------------------------------------------------------
__global__ __launch_bounds__(128) void graph_attn_kernel(
    const __bf16* __restrict__ qb, const __bf16* __restrict__ vt,
    const float* __restrict__ S, const float* __restrict__ ST,
    float* __restrict__ att, int N)
{
    __shared__ __align__(16) __bf16 qjs[32][LDPAD];   // q rows j0..j0+31
    __shared__ __align__(16) __bf16 vnT[LDPAD][32];   // vnT[c][k] = vn[j0+k][c]
    __shared__ __align__(16) float  Ssub[2][64][32];  // double buffered
    __shared__ __align__(16) float  STsub[2][64][32];
    __shared__ __align__(16) __bf16 wlds[4][16][32];  // per-wave C->A round-trip

    const int tid = threadIdx.x, wv = tid >> 5, l = tid & 31;
    const int i0 = blockIdx.x * 64;
    const int mcol = l & 15;
    const int hk = (l & 16) ? 8 : 0;
    const int bk = (l >> 4) << 4;
    const int hr = (l >> 4) << 3;

    // per-lane async mapping: wave wv stages rows [wv*16, wv*16+16) of S/ST
    const int srow = wv * 16 + (l >> 1);
    const int skb  = (l & 1) << 4;

    auto issueS = [&](int j0, int buf) {
        const float* s0 = S  + (size_t)(i0 + srow) * N + j0 + skb;
        const float* s1 = ST + (size_t)(i0 + srow) * N + j0 + skb;
        float* d0 = &Ssub[buf][srow][skb];
        float* d1 = &STsub[buf][srow][skb];
        #pragma unroll
        for (int p = 0; p < 4; ++p) {
            async_cp16(s0 + 4 * p, d0 + 4 * p);
            async_cp16(s1 + 4 * p, d1 + 4 * p);
        }
    };

    // Persistent A fragments of this wave's 16 q rows (K = 112, 4 chunks of 32)
    v16bf afr[4];
    {
        const __bf16* qrow = qb + (size_t)(i0 + wv * 16 + mcol) * LDPAD;
        #pragma unroll
        for (int kc = 0; kc < 4; ++kc) {
            v16bf a;
            #pragma unroll
            for (int i = 0; i < 16; ++i)
                a[i] = qrow[kc * 32 + bk + i];
            afr[kc] = a;
        }
    }
    v8f acc[7] = {};
    float rs[8] = {0.f, 0.f, 0.f, 0.f, 0.f, 0.f, 0.f, 0.f};

    issueS(0, 0);
    for (int j0 = 0; j0 < N; j0 += 32) {
        const int buf = (j0 >> 5) & 1;
        // ---- stage q / vn tiles: pure 16B copies (L2-resident operands) ----
        {
            const uint4* src = (const uint4*)(qb + (size_t)j0 * LDPAD);
            uint4* dst = (uint4*)&qjs[0][0];
            #pragma unroll
            for (int e = tid; e < (32 * LDPAD) / 8; e += 128) dst[e] = src[e];
        }
        for (int e = tid; e < 448; e += 128) {       // 112 cols x 4 chunks
            int c = e >> 2, p = e & 3;
            ((uint4*)&vnT[c][0])[p] = ((const uint4*)(vt + (size_t)c * N + j0))[p];
        }
        // ---- stream next S/ST tiles one iteration ahead ----
        if (j0 + 32 < N) { issueS(j0 + 32, buf ^ 1); ASYNC_WAIT(8); }
        else             { ASYNC_WAIT(0); }
        __syncthreads();

        // ---- score subtiles (16 j-cols each) + exp*mask, stash w in LDS ----
        #pragma unroll
        for (int s = 0; s < 2; ++s) {
            v8f sc = {};
            #pragma unroll
            for (int kc = 0; kc < 4; ++kc) {
                v16bf b;
                #pragma unroll
                for (int i = 0; i < 16; ++i) b[i] = qjs[16 * s + mcol][kc * 32 + bk + i];
                sc = __builtin_amdgcn_wmma_f32_16x16x32_bf16(false, afr[kc], false, b,
                                                             (short)0, sc, false, false);
            }
            #pragma unroll
            for (int r = 0; r < 8; ++r) {
                int mlcl = r + hr;                 // row within wave's 16-row tile
                int n = 16 * s + mcol;             // col within 32-wide j tile
                float w = expf(sc[r] * 0.1f)
                        * STsub[buf][wv * 16 + mlcl][n] * Ssub[buf][wv * 16 + mlcl][n];
                rs[r] += w;
                wlds[wv][mlcl][n] = (__bf16)w;
            }
        }
        __syncthreads();

        // ---- acc[16,112] += w[16,32] @ vn[32,112] ----
        v16bf aw;
        #pragma unroll
        for (int i = 0; i < 16; ++i) aw[i] = wlds[wv][mcol][bk + i];
        #pragma unroll
        for (int ct = 0; ct < 7; ++ct) {
            v16bf b;
            #pragma unroll
            for (int i = 0; i < 16; ++i) b[i] = vnT[ct * 16 + mcol][bk + i];
            acc[ct] = __builtin_amdgcn_wmma_f32_16x16x32_bf16(false, aw, false, b,
                                                              (short)0, acc[ct], false, false);
        }
        __syncthreads();
    }

    // reduce row sums across the 16 lanes of each half-wave, then normalize
    #pragma unroll
    for (int r = 0; r < 8; ++r)
        for (int o = 8; o; o >>= 1) rs[r] += __shfl_xor(rs[r], o, 32);

    #pragma unroll
    for (int ct = 0; ct < 7; ++ct) {
        int col = ct * 16 + mcol;
        if (col < 100) {
            #pragma unroll
            for (int r = 0; r < 8; ++r) {
                int row = i0 + wv * 16 + r + hr;
                att[(size_t)row * LDPAD + col] = acc[ct][r] / rs[r];
            }
        }
    }
}

// Final: BN(attended) -> relu(@ w_bm[100,30]) -> @ w_cox[30,1]. One wave/row.
__global__ __launch_bounds__(128) void final_cox_kernel(
    const float* __restrict__ att, const float* __restrict__ mean,
    const float* __restrict__ inv, const float* __restrict__ g,
    const float* __restrict__ be, const float* __restrict__ wbm,
    const float* __restrict__ wcox, float* __restrict__ out)
{
    __shared__ float h[4][100];
    const int wv = threadIdx.x >> 5, l = threadIdx.x & 31;
    const int row = blockIdx.x * 4 + wv;
    for (int c = l; c < 100; c += 32)
        h[wv][c] = g[c] * (att[(size_t)row * LDPAD + c] - mean[c]) * inv[c] + be[c];
    __syncthreads();
    float p = 0.f;
    if (l < 30) {
        float f = 0.f;
        for (int c = 0; c < 100; ++c) f += h[wv][c] * wbm[c * 30 + l];
        p = fmaxf(f, 0.f) * wcox[l];
    }
    p = wave_sum32(p);
    if (l == 0) out[row] = p;
}

// ---------------------------------------------------------------------------
extern "C" void kernel_launch(void* const* d_in, const int* in_sizes, int n_in,
                              void* d_out, int out_size, void* d_ws, size_t ws_size,
                              hipStream_t stream) {
    const float* gene   = (const float*)d_in[0];
    const float* miRNA  = (const float*)d_in[1];
    const float* S      = (const float*)d_in[2];
    const float* STr    = (const float*)d_in[3];
    const float* w_rna  = (const float*)d_in[4];
    const float* b_rna  = (const float*)d_in[5];
    const float* g_rna  = (const float*)d_in[6];
    const float* be_rna = (const float*)d_in[7];
    const float* w_mi   = (const float*)d_in[8];
    const float* b_mi   = (const float*)d_in[9];
    const float* g_mi   = (const float*)d_in[10];
    const float* be_mi  = (const float*)d_in[11];
    const float* wq1    = (const float*)d_in[12];
    const float* bq1    = (const float*)d_in[13];
    const float* wv1    = (const float*)d_in[14];
    const float* bv1    = (const float*)d_in[15];
    const float* wq2    = (const float*)d_in[16];
    const float* bq2    = (const float*)d_in[17];
    const float* wv2    = (const float*)d_in[18];
    const float* bv2    = (const float*)d_in[19];
    const float* w_fuse = (const float*)d_in[20];
    const float* b_fuse = (const float*)d_in[21];
    const float* g_bn   = (const float*)d_in[22];
    const float* be_bn  = (const float*)d_in[23];
    const float* wgq    = (const float*)d_in[24];
    const float* bgq    = (const float*)d_in[25];
    const float* wgv    = (const float*)d_in[26];
    const float* bgv    = (const float*)d_in[27];
    const float* g_gbn  = (const float*)d_in[28];
    const float* be_gbn = (const float*)d_in[29];
    const float* w_bm   = (const float*)d_in[30];
    const float* w_cox  = (const float*)d_in[31];

    float* ws = (float*)d_ws;
    const int N = NROWS;
    const size_t stride = (size_t)N * LDPAD;
    float* bGene  = ws;                 // [N,112] pre-BN then gene_f
    float* bMi    = ws + 1 * stride;    // [N,112] pre-BN then mi_f
    float* bFused = ws + 2 * stride;    // [N,112] fused
    float* bQ     = ws + 3 * stride;    // [N,112] graph q f32 (zero-padded)
    float* bV     = ws + 4 * stride;    // [N,112] graph v then l2n(v) (zero-padded)
    float* bAtt   = ws + 5 * stride;    // [N,112] attended
    float* sMean  = ws + 6 * stride;    // [128]
    float* sInv   = sMean + 128;        // [128]
    __bf16* qb = (__bf16*)(sInv + 128); // [N,112] bf16 q
    __bf16* vt = qb + stride;           // [112,N] bf16 l2n(v) transposed

    const dim3 gGemm(7, N / 64);
    const int bnApplyGrid = (N * 100 + 255) / 256;

    // --- omics encoders ---
    gemm_bias_kernel<<<gGemm, 128, 0, stream>>>(gene, 6000, w_rna, 100, b_rna,
                                                bGene, LDPAD, 6000, 100, (__bf16*)nullptr);
    bn_stats_kernel<<<100, 256, 0, stream>>>(bGene, N, sMean, sInv);
    bn_apply_kernel<<<bnApplyGrid, 256, 0, stream>>>(bGene, sMean, sInv, g_rna, be_rna, N, 1);

    gemm_bias_kernel<<<gGemm, 128, 0, stream>>>(miRNA, 600, w_mi, 100, b_mi,
                                                bMi, LDPAD, 600, 100, (__bf16*)nullptr);
    bn_stats_kernel<<<100, 256, 0, stream>>>(bMi, N, sMean, sInv);
    bn_apply_kernel<<<bnApplyGrid, 256, 0, stream>>>(bMi, sMean, sInv, g_mi, be_mi, N, 1);

    // --- two-token cooperative transformer + fuse ---
    token_attn_fuse_kernel<<<N / 4, 128, 0, stream>>>(bGene, bMi,
                                                      wq1, bq1, wv1, bv1,
                                                      wq2, bq2, wv2, bv2,
                                                      w_fuse, b_fuse, bFused);
    bn_stats_kernel<<<100, 256, 0, stream>>>(bFused, N, sMean, sInv);
    bn_apply_kernel<<<bnApplyGrid, 256, 0, stream>>>(bFused, sMean, sInv, g_bn, be_bn, N, 0);

    // --- graph q/v projections (q also emitted as bf16) ---
    gemm_bias_kernel<<<gGemm, 128, 0, stream>>>(bFused, LDPAD, wgq, 100, bgq,
                                                bQ, LDPAD, 100, 100, qb);
    gemm_bias_kernel<<<gGemm, 128, 0, stream>>>(bFused, LDPAD, wgv, 100, bgv,
                                                bV, LDPAD, 100, 100, (__bf16*)nullptr);
    l2norm_rows_kernel<<<N / 4, 128, 0, stream>>>(bV);
    build_vnT_kernel<<<dim3(N / 256, LDPAD), 256, 0, stream>>>(bV, vt, N);

    // --- fused graph attention (never materializes N x N) ---
    graph_attn_kernel<<<N / 64, 128, 0, stream>>>(qb, vt, S, STr, bAtt, N);

    // --- final head ---
    bn_stats_kernel<<<100, 256, 0, stream>>>(bAtt, N, sMean, sInv);
    final_cox_kernel<<<N / 4, 128, 0, stream>>>(bAtt, sMean, sInv, g_gbn, be_gbn,
                                                w_bm, w_cox, (float*)d_out);
}
